// GNNActorBase_16509854285899
// MI455X (gfx1250) — matrix operand order repaired and checked by reading, not compiled, not run on hardware
//
#include <hip/hip_runtime.h>

typedef __attribute__((ext_vector_type(16))) __bf16 v16bf;
typedef __attribute__((ext_vector_type(8)))  __bf16 v8bf;
typedef __attribute__((ext_vector_type(8)))  float  v8f;

#define N_IN 64
#define N_MID 256

// ---------------- elementwise / graph kernels ----------------

__global__ __launch_bounds__(256) void k_init(float* __restrict__ deg,
                                              float* __restrict__ agg, int n) {
  long tid = (long)blockIdx.x * blockDim.x + threadIdx.x;
  long total = (long)n * N_IN;
  if (tid < total) agg[tid] = 0.0f;
  if (tid < n) deg[tid] = 1.0f;  // self-loop contribution
}

__global__ __launch_bounds__(256) void k_deg(const int* __restrict__ dst,
                                             float* __restrict__ deg, int e, int n) {
  int tid = blockIdx.x * blockDim.x + threadIdx.x;
  if (tid >= e) return;
  unsigned d = (unsigned)dst[tid];
  if (d < (unsigned)n)
    __hip_atomic_fetch_add(&deg[d], 1.0f, __ATOMIC_RELAXED, __HIP_MEMORY_SCOPE_AGENT);
}

__global__ __launch_bounds__(256) void k_rsqrt(float* __restrict__ deg, int n) {
  int tid = blockIdx.x * blockDim.x + threadIdx.x;
  if (tid < n) deg[tid] = rsqrtf(deg[tid]);
}

__global__ __launch_bounds__(256) void k_cvt_bf16(const float* __restrict__ src,
                                                  __bf16* __restrict__ dst, long total) {
  long tid = (long)blockIdx.x * blockDim.x + threadIdx.x;
  if (tid < total) dst[tid] = (__bf16)src[tid];
}

// Pre-swizzle W[OUTC,K] (row-major) into per-lane-contiguous WMMA B fragments:
// Bswz[(((jt*(K/32)+kb)*32 + lane)*16 + i)] = bf16(W[j*K+k]),
//   j = jt*16 + (lane&15), k = kb*32 + (lane<16?0:8) + (i<8 ? i : i+8)
__global__ __launch_bounds__(256) void k_swizzle_w(const float* __restrict__ W,
                                                   __bf16* __restrict__ Bswz,
                                                   int K, int OUTC) {
  int tid = blockIdx.x * blockDim.x + threadIdx.x;
  if (tid >= K * OUTC) return;
  int i    = tid & 15;
  int lane = (tid >> 4) & 31;
  int rem  = tid >> 9;               // jt*(K/32)+kb
  int kb   = rem % (K / 32);
  int jt   = rem / (K / 32);
  int k = kb * 32 + ((lane < 16) ? 0 : 8) + ((i < 8) ? i : i + 8);
  int j = jt * 16 + (lane & 15);
  Bswz[tid] = (__bf16)W[(size_t)j * K + k];
}

// One wave per edge: agg[dst,:] += h[src,:] * dinv[src]*dinv[dst]
// Edge index made wave-uniform so src/dst/dinv become scalar loads.
__global__ __launch_bounds__(256) void k_scatter(const int* __restrict__ src,
                                                 const int* __restrict__ dst,
                                                 const float* __restrict__ h,
                                                 const float* __restrict__ dinv,
                                                 float* __restrict__ agg, int e, int n) {
  int lane = threadIdx.x & 31;
  int wslot = __builtin_amdgcn_readfirstlane(threadIdx.x >> 5);
  long wid = (long)blockIdx.x * 8 + wslot;
  if (wid >= e) return;
  unsigned s = (unsigned)src[wid], d = (unsigned)dst[wid];
  if (s >= (unsigned)n || d >= (unsigned)n) return;
  float norm = dinv[s] * dinv[d];
  const float2 hv = *(const float2*)(h + (size_t)s * N_IN + lane * 2);
  float* ap = agg + (size_t)d * N_IN + lane * 2;
  __hip_atomic_fetch_add(ap,     hv.x * norm, __ATOMIC_RELAXED, __HIP_MEMORY_SCOPE_AGENT);
  __hip_atomic_fetch_add(ap + 1, hv.y * norm, __ATOMIC_RELAXED, __HIP_MEMORY_SCOPE_AGENT);
}

// GCN epilogue: relu(agg + h*dinv^2 + conv_b) + x  -> bf16 residual activations
__global__ __launch_bounds__(256) void k_conv_epi(const float* __restrict__ agg,
                                                  const float* __restrict__ h,
                                                  const float* __restrict__ x,
                                                  const float* __restrict__ dinv,
                                                  const float* __restrict__ conv_b,
                                                  __bf16* __restrict__ hres, int n) {
  long tid = (long)blockIdx.x * blockDim.x + threadIdx.x;
  if (tid >= (long)n * N_IN) return;
  int node = (int)(tid >> 6);
  int c = (int)(tid & 63);
  float di = dinv[node];
  float v = agg[tid] + h[tid] * di * di + conv_b[c];
  v = fmaxf(v, 0.0f) + x[tid];
  hres[tid] = (__bf16)v;
}

// ---------------- WMMA GEMM: C[M,OUTC] = A[M,K] @ W^T (+bias, relu) ----------------
// One wave computes JT adjacent 16x16 column tiles of one 16-row block:
// A fragments loaded once per k-step, JT back-to-back WMMAs reuse them.

template <int K, int OUTC, int JT, bool BIAS_RELU, bool OUT_BF16>
__global__ __launch_bounds__(256) void k_gemm_wmma(const __bf16* __restrict__ A,
                                                   const __bf16* __restrict__ Bswz,
                                                   const float* __restrict__ bias,
                                                   float* __restrict__ Cf,
                                                   __bf16* __restrict__ Cb,
                                                   int nRows) {
  const int lane = threadIdx.x & 31;
  const int wave = blockIdx.x * 8 + (threadIdx.x >> 5);
  const int jGroups = OUTC / (16 * JT);
  const int mt = wave / jGroups;
  const int jt0 = (wave % jGroups) * JT;
  if (mt * 16 >= nRows) return;

  const int mRow = mt * 16 + (lane & 15);
  const int m = (mRow < nRows) ? mRow : (nRows - 1);
  const int jcol = lane & 15;
  const int aBase = (lane < 16) ? 0 : 8;  // ISA 16-bit A layout half-select

  v8f acc[JT];
#pragma unroll
  for (int t = 0; t < JT; ++t) {
    if (BIAS_RELU) {
      float bv = bias[(jt0 + t) * 16 + jcol];  // column N is jcol for all 8 rows
#pragma unroll
      for (int r = 0; r < 8; ++r) acc[t][r] = bv;
    } else {
      acc[t] = (v8f){};
    }
  }

  const __bf16* arow = A + (size_t)m * K;

#pragma unroll
  for (int kb = 0; kb < K / 32; ++kb) {
    union { v16bf v; v8bf h[2]; } a;
    // A: lanes 0-15 hold K = {0..7, 16..23}; lanes 16-31 hold K = {8..15, 24..31}
    a.h[0] = *(const v8bf*)(arow + kb * 32 + aBase);
    a.h[1] = *(const v8bf*)(arow + kb * 32 + 16 + aBase);
#pragma unroll
    for (int t = 0; t < JT; ++t) {
      union { v16bf v; v8bf h[2]; } b;
      // B: pre-swizzled so each lane's 16 values are contiguous
      const __bf16* bl =
          Bswz + ((((size_t)(jt0 + t) * (K / 32) + kb) * 32) + lane) * 16;
      b.h[0] = *(const v8bf*)(bl);
      b.h[1] = *(const v8bf*)(bl + 8);
      acc[t] = __builtin_amdgcn_wmma_f32_16x16x32_bf16(
          false, a.v, false, b.v, (short)0, acc[t], false, false);
    }
  }

  const int rowBase = mt * 16 + ((lane < 16) ? 0 : 8);
  if (rowBase + 8 <= nRows) {
    // full tile: branch-free clustered stores
#pragma unroll
    for (int t = 0; t < JT; ++t) {
      const int j = (jt0 + t) * 16 + jcol;
#pragma unroll
      for (int r = 0; r < 8; ++r) {
        float v = acc[t][r];
        if (BIAS_RELU) v = fmaxf(v, 0.0f);
        if (OUT_BF16) Cb[(size_t)(rowBase + r) * OUTC + j] = (__bf16)v;
        else          Cf[(size_t)(rowBase + r) * OUTC + j] = v;
      }
    }
  } else {
#pragma unroll
    for (int t = 0; t < JT; ++t) {
      const int j = (jt0 + t) * 16 + jcol;
#pragma unroll
      for (int r = 0; r < 8; ++r) {
        if (rowBase + r < nRows) {
          float v = acc[t][r];
          if (BIAS_RELU) v = fmaxf(v, 0.0f);
          if (OUT_BF16) Cb[(size_t)(rowBase + r) * OUTC + j] = (__bf16)v;
          else          Cf[(size_t)(rowBase + r) * OUTC + j] = v;
        }
      }
    }
  }
}

// Final [N,256] @ w3[1,256]^T + b3 : wave-per-node dot with shuffle reduction
__global__ __launch_bounds__(256) void k_final(const __bf16* __restrict__ m2,
                                               const float* __restrict__ w3,
                                               const float* __restrict__ b3,
                                               float* __restrict__ out, int n) {
  int lane = threadIdx.x & 31;
  int node = blockIdx.x * 8 + (threadIdx.x >> 5);
  if (node >= n) return;
  const __bf16* row = m2 + (size_t)node * N_MID;
  float s = 0.0f;
#pragma unroll
  for (int i = 0; i < N_MID / 32; ++i)
    s += (float)row[lane + i * 32] * w3[lane + i * 32];
#pragma unroll
  for (int off = 16; off > 0; off >>= 1) s += __shfl_xor(s, off, 32);
  if (lane == 0) out[node] = s + b3[0];
}

// ---------------- host launcher ----------------

extern "C" void kernel_launch(void* const* d_in, const int* in_sizes, int n_in,
                              void* d_out, int out_size, void* d_ws, size_t ws_size,
                              hipStream_t stream) {
  const float* x      = (const float*)d_in[0];
  const int*   eidx   = (const int*)d_in[1];
  const float* conv_w = (const float*)d_in[2];
  const float* conv_b = (const float*)d_in[3];
  const float* w1     = (const float*)d_in[4];
  const float* b1     = (const float*)d_in[5];
  const float* w2     = (const float*)d_in[6];
  const float* b2     = (const float*)d_in[7];
  const float* w3     = (const float*)d_in[8];
  const float* b3     = (const float*)d_in[9];
  float* out = (float*)d_out;

  const int n = in_sizes[0] / N_IN;      // 100000
  const int e = in_sizes[1] / 2;         // 1200000
  const int* src = eidx;
  const int* dst = eidx + e;

  // workspace carve-out (256B aligned)
  char* ws = (char*)d_ws;
  size_t off = 0;
  auto alloc = [&](size_t bytes) -> void* {
    size_t cur = (off + 255) & ~(size_t)255;
    off = cur + bytes;
    return (void*)(ws + cur);
  };
  float*  deg   = (float*)alloc((size_t)n * 4);
  float*  hlin  = (float*)alloc((size_t)n * N_IN * 4);   // x @ conv_w^T (f32)
  float*  agg   = (float*)alloc((size_t)n * N_IN * 4);   // scatter target
  __bf16* abf   = (__bf16*)alloc((size_t)n * N_IN * 2);  // x_bf16, reused as hres_bf16
  __bf16* m1b   = (__bf16*)alloc((size_t)n * N_MID * 2);
  __bf16* cwS   = (__bf16*)alloc((size_t)N_IN * N_IN * 2);
  __bf16* w1S   = (__bf16*)alloc((size_t)N_MID * N_IN * 2);
  __bf16* w2S   = (__bf16*)alloc((size_t)N_MID * N_MID * 2);
  // m2 (bf16, 51.2MB) overlays the dead hlin+agg region (contiguous 51.2MB)
  __bf16* m2b = (__bf16*)hlin;

  const long nc = (long)n * N_IN;
  const int B = 256;
  auto cdiv = [](long a, long b) { return (int)((a + b - 1) / b); };
  const long mTiles = (n + 15) / 16;

  // init deg (=1 self-loop) and zero agg
  k_init<<<cdiv(nc, B), B, 0, stream>>>(deg, agg, n);
  // degree accumulation + rsqrt
  k_deg<<<cdiv(e, B), B, 0, stream>>>(dst, deg, e, n);
  k_rsqrt<<<cdiv(n, B), B, 0, stream>>>(deg, n);
  // stage bf16 operands
  k_cvt_bf16<<<cdiv(nc, B), B, 0, stream>>>(x, abf, nc);
  k_swizzle_w<<<cdiv(N_IN * N_IN, B), B, 0, stream>>>(conv_w, cwS, N_IN, N_IN);
  k_swizzle_w<<<cdiv(N_MID * N_IN, B), B, 0, stream>>>(w1, w1S, N_IN, N_MID);
  k_swizzle_w<<<cdiv(N_MID * N_MID, B), B, 0, stream>>>(w2, w2S, N_MID, N_MID);

  // GEMM1: hlin = x @ conv_w^T   (no bias/relu, f32 out, 2 col-tiles/wave)
  {
    long waves = mTiles * (N_IN / 16 / 2);
    k_gemm_wmma<N_IN, N_IN, 2, false, false>
        <<<cdiv(waves, 8), B, 0, stream>>>(abf, cwS, nullptr, hlin, nullptr, n);
  }
  // graph aggregation
  k_scatter<<<cdiv((long)e, 8), B, 0, stream>>>(src, dst, hlin, deg, agg, e, n);
  // epilogue: relu(agg + self-loop + bias) + x -> bf16 (reuses abf)
  k_conv_epi<<<cdiv(nc, B), B, 0, stream>>>(agg, hlin, x, deg, conv_b, abf, n);

  // GEMM2: m1 = relu(hres @ w1^T + b1)  (bf16 out, 4 col-tiles/wave)
  {
    long waves = mTiles * (N_MID / 16 / 4);
    k_gemm_wmma<N_IN, N_MID, 4, true, true>
        <<<cdiv(waves, 8), B, 0, stream>>>(abf, w1S, b1, nullptr, m1b, n);
  }
  // GEMM3: m2 = relu(m1 @ w2^T + b2)  (bf16 out, overlays hlin/agg)
  {
    long waves = mTiles * (N_MID / 16 / 4);
    k_gemm_wmma<N_MID, N_MID, 4, true, true>
        <<<cdiv(waves, 8), B, 0, stream>>>(m1b, w2S, b2, nullptr, m2b, n);
  }
  // final projection [N,256] -> [N,1]
  k_final<<<cdiv((long)n, 8), B, 0, stream>>>(m2b, w3, b3, out, n);
}
